// Wrapped8BitLinear_26053271617745
// MI455X (gfx1250) — compile-verified, matrix operand order checked by reading, not run on hardware
//
#include <hip/hip_runtime.h>
#include <stdint.h>

// Problem constants (from the reference): M = B*S, K = D_IN, N = D_OUT
#define M_DIM 8192
#define N_DIM 11008
#define K_DIM 4096

#define BM 128
#define BN 128
#define BK 32
#define LDSK 40   // padded row stride in halves (80 bytes, 16B aligned)

typedef _Float16 h8   __attribute__((ext_vector_type(8)));
typedef _Float16 v16h __attribute__((ext_vector_type(16)));
typedef float    v8f  __attribute__((ext_vector_type(8)));
typedef int      i4v  __attribute__((ext_vector_type(4)));
typedef int      i4g  __attribute__((vector_size(16)));   // matches builtin's V4i

// ---- CDNA5 async global->LDS copy (16B), with portable fallback ----
__device__ __forceinline__ void async_b128(const _Float16* g, _Float16* l) {
#if __has_builtin(__builtin_amdgcn_global_load_async_to_lds_b128)
  __builtin_amdgcn_global_load_async_to_lds_b128(
      (__attribute__((address_space(1))) i4g*)(unsigned long long)(uintptr_t)g,
      (__attribute__((address_space(3))) i4g*)(unsigned)(uintptr_t)l,
      0, 0);
#else
  *(h8*)l = *(const h8*)g;
#endif
}

__device__ __forceinline__ void wait_async0() {
#if __has_builtin(__builtin_amdgcn_s_wait_asynccnt)
  __builtin_amdgcn_s_wait_asynccnt(0);
#else
  asm volatile("s_wait_asynccnt 0" ::: "memory");
#endif
}

__global__ __launch_bounds__(256)
void w8a16_linear_wmma(const _Float16* __restrict__ x,    // [M, K] fp16
                       const int*      __restrict__ CB,   // [N, K] int8 in int32
                       const _Float16* __restrict__ SCB,  // [N] fp16
                       const _Float16* __restrict__ bias, // [N] fp16
                       float*          __restrict__ out)  // [M, N] fp32
{
  __shared__ __align__(16) _Float16 As[2][BM][LDSK];
  __shared__ __align__(16) _Float16 Bs[2][BN][LDSK];

  const int t      = threadIdx.x;
  const int m_base = blockIdx.y * BM;
  const int n_base = blockIdx.x * BN;

  // tile-fill mapping: each thread owns 16 consecutive K elements of one row
  const int lrow = t >> 1;          // 0..127
  const int lseg = (t & 1) << 4;    // 0 or 16 (halves)

  // wave/lane mapping for WMMA fragments
  const int wave = t >> 5;
  const int lane = t & 31;
  const int wm   = wave >> 1;       // 0..3 : 32-row M band
  const int wn   = wave & 1;        // 0..1 : 64-col N band
  const int fm   = lane & 15;       // row/col within 16x16 fragment
  const int half = lane >> 4;       // selects K half per ISA VGPR layout

  v8f acc[2][4] = {};

  auto load_tile = [&](int buf, int kk) {
    // A tile: 128x32 fp16, async DMA straight into LDS (2 x b128 per thread)
    const _Float16* gA = x + (size_t)(m_base + lrow) * K_DIM + kk + lseg;
    async_b128(gA,     &As[buf][lrow][lseg]);
    async_b128(gA + 8, &As[buf][lrow][lseg + 8]);

    // B tile: 128x32 int32 -> exact fp16 convert -> LDS (scale folded into epilogue)
    const i4v* gB = (const i4v*)(CB + (size_t)(n_base + lrow) * K_DIM + kk + lseg);
    i4v w0 = gB[0], w1 = gB[1], w2 = gB[2], w3 = gB[3];
    h8 p0 = { (_Float16)w0.x, (_Float16)w0.y, (_Float16)w0.z, (_Float16)w0.w,
              (_Float16)w1.x, (_Float16)w1.y, (_Float16)w1.z, (_Float16)w1.w };
    h8 p1 = { (_Float16)w2.x, (_Float16)w2.y, (_Float16)w2.z, (_Float16)w2.w,
              (_Float16)w3.x, (_Float16)w3.y, (_Float16)w3.z, (_Float16)w3.w };
    *(h8*)&Bs[buf][lrow][lseg]     = p0;
    *(h8*)&Bs[buf][lrow][lseg + 8] = p1;
  };

  load_tile(0, 0);

  const int KT = K_DIM / BK;  // 128
  for (int kt = 0; kt < KT; ++kt) {
    const int buf = kt & 1;
    wait_async0();
    __syncthreads();                         // tile `buf` ready for everyone
    if (kt + 1 < KT) load_tile(buf ^ 1, (kt + 1) * BK);  // overlap next fill

    // A fragments (16x32 fp16): two b128 LDS reads per fragment, per ISA layout
    v16h afrag[2];
#pragma unroll
    for (int mi = 0; mi < 2; ++mi) {
      const _Float16* ap = &As[buf][32 * wm + 16 * mi + fm][0];
      h8 lo = *(const h8*)(ap + 8 * half);
      h8 hi = *(const h8*)(ap + 16 + 8 * half);
      afrag[mi] = __builtin_shufflevector(lo, hi,
                    0,1,2,3,4,5,6,7,8,9,10,11,12,13,14,15);
    }
#pragma unroll
    for (int ni = 0; ni < 4; ++ni) {
      const _Float16* bp = &Bs[buf][64 * wn + 16 * ni + fm][0];
      h8 lo = *(const h8*)(bp + 8 * half);
      h8 hi = *(const h8*)(bp + 16 + 8 * half);
      v16h bfrag = __builtin_shufflevector(lo, hi,
                     0,1,2,3,4,5,6,7,8,9,10,11,12,13,14,15);
#pragma unroll
      for (int mi = 0; mi < 2; ++mi) {
        acc[mi][ni] = __builtin_amdgcn_wmma_f32_16x16x32_f16(
            false, afrag[mi], false, bfrag, (short)0, acc[mi][ni], false, false);
      }
    }
    __syncthreads();   // everyone done with `buf` before it is refilled next iter
  }

  // Epilogue: out = acc * (SCB[n]/127) + bias[n]   (scale constant per N column)
#pragma unroll
  for (int ni = 0; ni < 4; ++ni) {
    const int n = n_base + 64 * wn + 16 * ni + fm;
    const float sc = (float)SCB[n] * (1.0f / 127.0f);
    const float bi = (float)bias[n];
#pragma unroll
    for (int mi = 0; mi < 2; ++mi) {
      const int m0 = m_base + 32 * wm + 16 * mi + 8 * half;
      float* op = out + (size_t)m0 * N_DIM + n;
#pragma unroll
      for (int r = 0; r < 8; ++r) {
        op[(size_t)r * N_DIM] = acc[mi][ni][r] * sc + bi;
      }
    }
  }
}

extern "C" void kernel_launch(void* const* d_in, const int* in_sizes, int n_in,
                              void* d_out, int out_size, void* d_ws, size_t ws_size,
                              hipStream_t stream) {
  (void)in_sizes; (void)n_in; (void)out_size; (void)d_ws; (void)ws_size;
  const _Float16* x    = (const _Float16*)d_in[0];
  const int*      CB   = (const int*)d_in[1];
  const _Float16* SCB  = (const _Float16*)d_in[2];
  const _Float16* bias = (const _Float16*)d_in[3];
  float*          out  = (float*)d_out;

  dim3 grid(N_DIM / BN, M_DIM / BM);   // 86 x 64 blocks
  w8a16_linear_wmma<<<grid, 256, 0, stream>>>(x, CB, SCB, bias, out);
}
